// Linear_53961969107198
// MI455X (gfx1250) — compile-verified, matrix operand order checked by reading, not run on hardware
//
#include <hip/hip_runtime.h>

// CDNA5 / gfx1250: wave32, WMMA 16x16x32 bf16 (f32 accum), TDM tensor load.
// Segmented irrep linear: out_seg = 128^-0.5 * W_seg^T @ x_seg (+ bias on seg 0).
// bf16x3 split-precision (hi/lo) => ~fp32 accuracy while staying HBM-bound:
//   922 MB mandatory traffic @ 23.3 TB/s ~= 40us; 3x-emulated bf16 WMMA needs
//   ~96 FLOP/byte vs a bf16 machine balance of ~200+ => memory bound.

typedef __attribute__((ext_vector_type(16))) __bf16 v16bf;
typedef __attribute__((ext_vector_type(8)))  float  v8f;
typedef __attribute__((ext_vector_type(4)))  unsigned int u32x4;
typedef __attribute__((ext_vector_type(8)))  int          i32x8;
typedef __attribute__((ext_vector_type(4)))  int          i32x4;

#define NT   32      // nodes per workgroup tile
#define DIMT 1152    // total feature dim (128*1 + 128*3 + 128*5)
#define MUL  128     // multiplicity (K and V dimension of each segment GEMM)

static __device__ __forceinline__ unsigned short f2bf(float f) {
  union { float f; unsigned u; } a; a.f = f;
  unsigned r = a.u + 0x7fffu + ((a.u >> 16) & 1u);   // round-to-nearest-even
  return (unsigned short)(r >> 16);
}
static __device__ __forceinline__ float bf2f(unsigned short h) {
  union { unsigned u; float f; } a; a.u = ((unsigned)h) << 16; return a.f;
}

union BfFrag { v16bf v; unsigned u[8]; };

// LDS layout (u-fast, padded rows: stride 65 dwords == 1 mod 64 banks):
//   xhi/xlo : [c = nn*D + i][u]  bf16, C x 130
//   whi/wlo : [v][u]  (= W^T)    bf16, 128 x 130, pre-scaled by 128^-0.5
//   outs    : [c][v]             f32,  C x 129  (out staging, coalesced store)
//   rawx    : aliases outs       f32,  NT x 128*D raw x tile (TDM destination;
//             dead after the conversion barrier, before outs goes live)
template<int D, bool BIAS>
__global__ __launch_bounds__(256) void seg_linear(
    const float* __restrict__ x, const float* __restrict__ w,
    const float* __restrict__ bias, float* __restrict__ out,
    int n_nodes, int xoff, int woff)
{
  constexpr int C  = NT * D;
  constexpr int XS = 130;   // halves
  constexpr int WS = 130;   // halves
  constexpr int OS = 129;   // floats
  constexpr int ROWE = MUL * D;                 // floats per node-row
  constexpr int ELEMS = NT * ROWE;
  constexpr unsigned OUTS_OFF = 2u * C * XS * 2u + 2u * MUL * WS * 2u; // bytes
  constexpr float PW = 0.08838834764831845f;    // 128^-0.5

  extern __shared__ unsigned char smem_raw[];
  unsigned short* xhi = (unsigned short*)smem_raw;
  unsigned short* xlo = xhi + C * XS;
  unsigned short* whi = xlo + C * XS;
  unsigned short* wlo = whi + MUL * WS;
  float*          outs = (float*)(smem_raw + OUTS_OFF);
  float*          rawx = outs;                  // intentional alias (see above)

  const int t     = threadIdx.x;
  const int node0 = blockIdx.x * NT;

  // ---- Phase 0: kick off TDM DMA of the raw x tile into LDS (wave 0 only;
  // TDM ignores EXEC, issues once per wave). 2D tile: 128*D floats x NT nodes,
  // row stride DIMT floats; rows past n_nodes zero-filled by tensor OOB rules.
#if __has_builtin(__builtin_amdgcn_tensor_load_to_lds)
  constexpr bool USE_TDM = true;
  if (t < 32) {
    const unsigned long long ga =
        (unsigned long long)(const void*)(x + (size_t)node0 * DIMT + xoff);
    const unsigned td1 = (unsigned)(n_nodes > node0 ? n_nodes - node0 : 0);
    u32x4 g0;
    g0[0] = 1u;                                   // count=1 (valid), user mode
    g0[1] = OUTS_OFF;                             // lds_addr (bytes)
    g0[2] = (unsigned)(ga & 0xffffffffu);         // global_addr[31:0]
    g0[3] = (unsigned)((ga >> 32) & 0x1ffffffu)   // global_addr[56:32]
          | 0x80000000u;                          // type=2 ("image")
    i32x8 g1;
    g1[0] = (int)(2u << 16);                      // data_size=4B, wg_mask=0
    g1[1] = (int)((ROWE & 0xffffu) << 16);        // tensor_dim0[15:0]
    g1[2] = (int)(((unsigned)ROWE >> 16) | ((td1 & 0xffffu) << 16));
    g1[3] = (int)((td1 >> 16) | ((unsigned)ROWE << 16));   // tile_dim0
    g1[4] = (int)NT;                              // tile_dim1=NT, tile_dim2=0
    g1[5] = (int)DIMT;                            // tensor_dim0_stride[31:0]
    g1[6] = 0;
    g1[7] = 0;
    i32x4 gz4 = {0, 0, 0, 0};
    i32x8 gz8 = {0, 0, 0, 0, 0, 0, 0, 0};
    __builtin_amdgcn_tensor_load_to_lds(g0, g1, gz4, gz4, gz8, 0);
  }
#else
  constexpr bool USE_TDM = false;
#endif

  // ---- Phase 1 (overlaps DMA): W^T (scaled) -> LDS bf16 hi/lo.
  // w[u*128+v] -> row v, col u.  B128 loads: 4 consecutive v, same u.
  for (int e4 = t * 4; e4 < MUL * MUL; e4 += 1024) {
    const float4 f = *(const float4*)(w + woff + e4);
    const int u = e4 >> 7;
    #pragma unroll
    for (int q = 0; q < 4; ++q) {
      const int v = (e4 & (MUL - 1)) + q;
      const float fv = (&f.x)[q] * PW;
      const unsigned short hi = f2bf(fv);
      whi[v * WS + u] = hi;
      wlo[v * WS + u] = f2bf(fv - bf2f(hi));
    }
  }

#if __has_builtin(__builtin_amdgcn_tensor_load_to_lds)
  if (t < 32) __builtin_amdgcn_s_wait_tensorcnt((short)0);
#endif
  __syncthreads();   // raw x tile visible to all waves

  // ---- Phase 2: raw x tile -> bf16 hi/lo, transposing (u,i) -> row c, col u.
  for (int e4 = t * 4; e4 < ELEMS; e4 += 1024) {
    float4 f;
    if (USE_TDM) {
      f = *(const float4*)(rawx + e4);
    } else {
      const int nn = e4 / ROWE;
      const int n  = node0 + nn;
      if (n < n_nodes) f = *(const float4*)(x + (size_t)n * DIMT + xoff + (e4 - nn * ROWE));
      else             f = make_float4(0.f, 0.f, 0.f, 0.f);
    }
    #pragma unroll
    for (int q = 0; q < 4; ++q) {
      const int e  = e4 + q;
      const int nn = e / ROWE;
      const int r  = e - nn * ROWE;
      const int u  = r / D;
      const int i  = r - u * D;
      const float fv = (&f.x)[q];
      const unsigned short hi = f2bf(fv);
      xhi[(nn * D + i) * XS + u] = hi;
      xlo[(nn * D + i) * XS + u] = f2bf(fv - bf2f(hi));
    }
  }
  __syncthreads();   // bf16 tiles ready; rawx region is now dead -> outs

  // ---- Phase 3: WMMA. Wave wv owns v-rows [(wv&3)*32, +32) (two 16-row
  // tiles, A fragments resident in VGPRs) and strides over 16-col c-tiles.
  const int wv  = t >> 5;
  const int ln  = t & 31;
  const int lh  = ln >> 4;
  const int l15 = ln & 15;
  const int v0  = (wv & 3) * 32;

  unsigned a_hi[2][4][8], a_lo[2][4][8];
  #pragma unroll
  for (int vt = 0; vt < 2; ++vt) {
    const int vrow = v0 + vt * 16 + l15;
    #pragma unroll
    for (int kc = 0; kc < 4; ++kc) {
      #pragma unroll
      for (int j = 0; j < 8; ++j) {
        const int u = kc * 32 + ((j < 4) ? 2 * j : 2 * j + 8) + lh * 8;
        a_hi[vt][kc][j] = *(const unsigned*)(whi + vrow * WS + u);
        a_lo[vt][kc][j] = *(const unsigned*)(wlo + vrow * WS + u);
      }
    }
  }

  for (int ct = (wv >> 2); ct < 2 * D; ct += 2) {
    const int ccol = ct * 16 + l15;
    v8f acc0 = {};
    v8f acc1 = {};
    #pragma unroll
    for (int kc = 0; kc < 4; ++kc) {
      BfFrag Bh, Bl, Ah0, Al0, Ah1, Al1;
      #pragma unroll
      for (int j = 0; j < 8; ++j) {
        const int u = kc * 32 + ((j < 4) ? 2 * j : 2 * j + 8) + lh * 8;
        Bh.u[j]  = *(const unsigned*)(xhi + ccol * XS + u);
        Bl.u[j]  = *(const unsigned*)(xlo + ccol * XS + u);
        Ah0.u[j] = a_hi[0][kc][j];  Al0.u[j] = a_lo[0][kc][j];
        Ah1.u[j] = a_hi[1][kc][j];  Al1.u[j] = a_lo[1][kc][j];
      }
      acc0 = __builtin_amdgcn_wmma_f32_16x16x32_bf16(false, Ah0.v, false, Bh.v, (short)0, acc0, false, false);
      acc1 = __builtin_amdgcn_wmma_f32_16x16x32_bf16(false, Ah1.v, false, Bh.v, (short)0, acc1, false, false);
      acc0 = __builtin_amdgcn_wmma_f32_16x16x32_bf16(false, Ah0.v, false, Bl.v, (short)0, acc0, false, false);
      acc1 = __builtin_amdgcn_wmma_f32_16x16x32_bf16(false, Ah1.v, false, Bl.v, (short)0, acc1, false, false);
      acc0 = __builtin_amdgcn_wmma_f32_16x16x32_bf16(false, Al0.v, false, Bh.v, (short)0, acc0, false, false);
      acc1 = __builtin_amdgcn_wmma_f32_16x16x32_bf16(false, Al1.v, false, Bh.v, (short)0, acc1, false, false);
    }
    // D layout: VGPR r -> M = r (+8 in upper lane half), N = l15.
    #pragma unroll
    for (int r = 0; r < 8; ++r) {
      const int va = v0 +  0 + lh * 8 + r;
      const int vb = v0 + 16 + lh * 8 + r;
      float f0 = acc0[r], f1 = acc1[r];
      if (BIAS) { f0 += bias[va]; f1 += bias[vb]; }
      outs[ccol * OS + va] = f0;
      outs[ccol * OS + vb] = f1;
    }
  }
  __syncthreads();

  // ---- Phase 4: coalesced B128 store, un-transposing [c][v] -> out[n, v*D+i].
  for (int e4 = t * 4; e4 < ELEMS; e4 += 1024) {
    const int nn = e4 / ROWE;
    const int n  = node0 + nn;
    if (n >= n_nodes) continue;
    const int r0 = e4 - nn * ROWE;
    float4 o;
    #pragma unroll
    for (int q = 0; q < 4; ++q) {
      const int r = r0 + q;
      const int v = r / D;
      const int i = r - v * D;
      (&o.x)[q] = outs[(nn * D + i) * OS + v];
    }
    *(float4*)(out + (size_t)n * DIMT + xoff + r0) = o;
  }
}

static inline size_t smem_bytes(int D) {
  const size_t C = (size_t)NT * D;
  return C * 130 * 2 * 2            // xhi + xlo
       + (size_t)MUL * 130 * 2 * 2  // whi + wlo
       + C * 129 * 4;               // out staging (aliases raw TDM x tile)
}

extern "C" void kernel_launch(void* const* d_in, const int* in_sizes, int n_in,
                              void* d_out, int out_size, void* d_ws, size_t ws_size,
                              hipStream_t stream) {
  const float* x = (const float*)d_in[0];
  const float* w = (const float*)d_in[1];
  const float* b = (const float*)d_in[2];
  float* out = (float*)d_out;
  const int n_nodes = in_sizes[0] / DIMT;
  const int nblk = (n_nodes + NT - 1) / NT;

  (void)hipFuncSetAttribute((const void*)seg_linear<1, true >,
                      hipFuncAttributeMaxDynamicSharedMemorySize, (int)smem_bytes(1));
  (void)hipFuncSetAttribute((const void*)seg_linear<3, false>,
                      hipFuncAttributeMaxDynamicSharedMemorySize, (int)smem_bytes(3));
  (void)hipFuncSetAttribute((const void*)seg_linear<5, false>,
                      hipFuncAttributeMaxDynamicSharedMemorySize, (int)smem_bytes(5));

  // Segment offsets: x/out float offsets {0,128,512}; w offsets {0,16384,32768}.
  seg_linear<1, true ><<<nblk, 256, smem_bytes(1), stream>>>(x, w, b, out, n_nodes, 0,   0);
  seg_linear<3, false><<<nblk, 256, smem_bytes(3), stream>>>(x, w, b, out, n_nodes, 128, 16384);
  seg_linear<5, false><<<nblk, 256, smem_bytes(5), stream>>>(x, w, b, out, n_nodes, 512, 32768);
}
